// ParallelPalmTransformerLayer_23596550324231
// MI455X (gfx1250) — matrix operand order repaired and checked by reading, not compiled, not run on hardware
//
#include <hip/hip_runtime.h>
#include <hip/hip_bf16.h>
#include <cstdint>

typedef __bf16 bf16_t;
typedef __attribute__((ext_vector_type(16))) __bf16 v16bf;
typedef __attribute__((ext_vector_type(8)))  float  v8f;

#define DIM 2048
#define S_LEN 2048
#define IN_WIDTH 22528
#define USED_WIDTH 14336   /* 3*ATTN_INNER + FFN_INNER; rest of W_in is unused */
#define ATTN_INNER 2048
#define FFN_INNER 8192
#define OUT_IN 10240
#define NUM_HEADS 32
#define DIM_HEAD 64

union Frag { v16bf v; uint4 q[2]; };

__device__ __forceinline__ v16bf lds_frag(const char* base, int off0, int off1) {
  Frag f;
  f.q[0] = *(const uint4*)(base + off0);
  f.q[1] = *(const uint4*)(base + off1);
  return f.v;
}

// Async global->LDS 32-byte copy (two b128). IOFFSET applies to BOTH the LDS
// and the global address (ISA 15.18.3), so one LDS base VGPR serves both.
__device__ __forceinline__ void async_copy_32B(uint32_t lds_addr, const void* gaddr) {
  asm volatile("global_load_async_to_lds_b128 %0, %1, off"
               :: "v"(lds_addr), "v"(gaddr) : "memory");
  asm volatile("global_load_async_to_lds_b128 %0, %1, off offset:16"
               :: "v"(lds_addr), "v"(gaddr) : "memory");
}

__device__ __forceinline__ void wait_async0() {
  asm volatile("s_wait_asynccnt 0x0" ::: "memory");
}

// ---------------------------------------------------------------------------
// LayerNorm over rows of x (2048 cols), output bf16
// ---------------------------------------------------------------------------
__global__ __launch_bounds__(256) void ln_kernel(const float* __restrict__ x,
                                                 const float* __restrict__ w,
                                                 const float* __restrict__ b,
                                                 bf16_t* __restrict__ h) {
  const int row = blockIdx.x, tid = threadIdx.x;
  const float* xr = x + (size_t)row * DIM;
  float vals[8], s = 0.f, ss = 0.f;
#pragma unroll
  for (int i = 0; i < 8; ++i) {
    float vv = xr[tid + i * 256];
    vals[i] = vv; s += vv; ss += vv * vv;
  }
#pragma unroll
  for (int m = 16; m >= 1; m >>= 1) {
    s  += __shfl_xor(s,  m, 32);
    ss += __shfl_xor(ss, m, 32);
  }
  __shared__ float red[2][8];
  if ((tid & 31) == 0) { red[0][tid >> 5] = s; red[1][tid >> 5] = ss; }
  __syncthreads();
  if (tid == 0) {
    float ts = 0.f, tss = 0.f;
    for (int i = 0; i < 8; ++i) { ts += red[0][i]; tss += red[1][i]; }
    red[0][0] = ts; red[1][0] = tss;
  }
  __syncthreads();
  const float mean = red[0][0] * (1.f / DIM);
  const float var  = red[1][0] * (1.f / DIM) - mean * mean;
  const float rstd = rsqrtf(var + 1e-5f);
#pragma unroll
  for (int i = 0; i < 8; ++i) {
    const int c = tid + i * 256;
    h[(size_t)row * DIM + c] = (bf16_t)((vals[i] - mean) * rstd * w[c] + b[c]);
  }
}

// ---------------------------------------------------------------------------
// bf16 WMMA GEMM: C[M,N] = A[M,K](bf16) * B[K,N](f32, converted in staging)
// A may be a concat of two sources split at column a_split.
// Output either bf16 (Cbf) or f32 (Cf).
// Block tile 128x128, K-step 32, 256 threads = 8 waves (2x4 tiles each).
// A tile staged via GLOBAL_LOAD_ASYNC_TO_LDS_B128 (pure bf16 copy);
// B tile staged through VGPRs (f32 -> packed bf16, transposed [n][k]).
// ---------------------------------------------------------------------------
__global__ __launch_bounds__(256) void gemm_bf16_kernel(
    const bf16_t* __restrict__ A0, int lda0,
    const bf16_t* __restrict__ A1, int lda1, int a_split,
    const float* __restrict__ B, int ldb,
    bf16_t* __restrict__ Cbf, float* __restrict__ Cf, int ldc, int K) {
  __shared__ __align__(16) bf16_t ldsA[128 * 40];   // row-major, pad stride 40
  __shared__ __align__(16) bf16_t ldsB[128 * 40];   // transposed [n][k], pad 40

  const int tid  = threadIdx.x;
  const int lane = tid & 31, w = tid >> 5;
  const int wr = w >> 1, wc = w & 1;           // wave grid 4x2 over 128x128
  const int mbase = blockIdx.y * 128, nbase = blockIdx.x * 128;
  const int l15 = lane & 15, hi = lane >> 4;

  v8f acc[2][4] = {};

  const int arow = tid >> 1, ahalf = tid & 1;        // A: 128 rows x 32 cols
  const uint32_t aLds = (uint32_t)(uintptr_t)&ldsA[0] + (uint32_t)(arow * 80 + ahalf * 32);
  const int bn8 = (tid & 15) * 8, bk2 = (tid >> 4) * 2;  // B: k-pair x 8 n

  for (int kb = 0; kb < K; kb += 32) {
    __syncthreads();
    {  // stage A tile: async global->LDS (possibly concat source)
      const int col0 = kb + ahalf * 16;
      const bf16_t* src; int colrel;
      if (col0 < a_split) { src = A0 + (size_t)(mbase + arow) * lda0; colrel = col0; }
      else                { src = A1 + (size_t)(mbase + arow) * lda1; colrel = col0 - a_split; }
      async_copy_32B(aLds, src + colrel);
    }
    {  // stage B tile: f32 load, cvt to packed bf16 pairs (k,k+1), transpose
      const float* src = B + (size_t)(kb + bk2) * ldb + nbase + bn8;
      float4 r0a = ((const float4*)src)[0];
      float4 r0b = ((const float4*)src)[1];
      float4 r1a = ((const float4*)(src + ldb))[0];
      float4 r1b = ((const float4*)(src + ldb))[1];
      const float k0[8] = {r0a.x, r0a.y, r0a.z, r0a.w, r0b.x, r0b.y, r0b.z, r0b.w};
      const float k1[8] = {r1a.x, r1a.y, r1a.z, r1a.w, r1b.x, r1b.y, r1b.z, r1b.w};
#pragma unroll
      for (int i = 0; i < 8; ++i) {
        union { bf16_t hh[2]; uint32_t u; } pk;
        pk.hh[0] = (bf16_t)k0[i];
        pk.hh[1] = (bf16_t)k1[i];
        *(uint32_t*)&ldsB[(bn8 + i) * 40 + bk2] = pk.u;
      }
      if (kb + 32 < K)
        __builtin_prefetch(B + (size_t)(kb + 32 + bk2) * ldb + nbase + bn8, 0, 1);
    }
    wait_async0();
    __syncthreads();

    v16bf af[2], bfr[4];
#pragma unroll
    for (int rt = 0; rt < 2; ++rt) {
      const int row = wr * 32 + rt * 16 + l15;
      const char* p = (const char*)&ldsA[row * 40] + hi * 16;
      af[rt] = lds_frag(p, 0, 32);   // K 0-7/8-15 then 16-23/24-31 per half-wave
    }
#pragma unroll
    for (int ct = 0; ct < 4; ++ct) {
      const int n = wc * 64 + ct * 16 + l15;
      const char* p = (const char*)&ldsB[n * 40] + hi * 16;
      bfr[ct] = lds_frag(p, 0, 32);  // K 0-7/8-15 then 16-23/24-31 per half-wave
    }
#pragma unroll
    for (int rt = 0; rt < 2; ++rt)
#pragma unroll
      for (int ct = 0; ct < 4; ++ct)
        acc[rt][ct] = __builtin_amdgcn_wmma_f32_16x16x32_bf16(
            false, af[rt], false, bfr[ct], (short)0, acc[rt][ct], false, false);
  }

#pragma unroll
  for (int rt = 0; rt < 2; ++rt)
#pragma unroll
    for (int ct = 0; ct < 4; ++ct) {
      const int col = nbase + wc * 64 + ct * 16 + l15;
#pragma unroll
      for (int r = 0; r < 8; ++r) {
        const int row = mbase + wr * 32 + rt * 16 + hi * 8 + r;
        const float vv = acc[rt][ct][r];
        if (Cbf) Cbf[(size_t)row * ldc + col] = (bf16_t)vv;
        else     Cf[(size_t)row * ldc + col]  = vv;
      }
    }
}

// ---------------------------------------------------------------------------
// Flash attention (causal). One block = (head, 128-query tile), 8 waves each
// owning a 16-row strip. Key blocks of 64 with online softmax.
// K tile staged async (pure copy); V tile transposed through VGPRs.
// ---------------------------------------------------------------------------
__global__ __launch_bounds__(256) void attn_kernel(const bf16_t* __restrict__ res,
                                                   bf16_t* __restrict__ attn_out) {
  __shared__ __align__(16) bf16_t Kt[64 * 72];       // [key][d], stride 72
  __shared__ __align__(16) bf16_t Vt[64 * 72];       // [d][key], stride 72
  __shared__ __align__(16) bf16_t Pst[8][16 * 72];   // per-wave P strip

  const int head = blockIdx.x, qt = blockIdx.y;
  const int qbase = qt * 128;
  const int tid = threadIdx.x, lane = tid & 31, w = tid >> 5;
  const int l15 = lane & 15, hi = lane >> 4;
  const float SL2 = 0.125f * 1.4426950408889634f;    // DIM_HEAD^-0.5 * log2(e)

  const bf16_t* qp = res + head * DIM_HEAD;
  const bf16_t* kp = res + ATTN_INNER + head * DIM_HEAD;
  const bf16_t* vp = res + 2 * ATTN_INNER + head * DIM_HEAD;

  // Q fragments for this wave's 16 rows (loaded once)
  const int qrow = qbase + w * 16 + l15;
  v16bf qf[2];
#pragma unroll
  for (int ks = 0; ks < 2; ++ks) {
    const bf16_t* p = qp + (size_t)qrow * USED_WIDTH + ks * 32 + hi * 8;
    Frag f; f.q[0] = *(const uint4*)p; f.q[1] = *(const uint4*)(p + 16);
    qf[ks] = f.v;
  }

  float mrow[8], lrow[8];
  v8f o[4] = {};
#pragma unroll
  for (int r = 0; r < 8; ++r) { mrow[r] = -3.0e38f; lrow[r] = 0.f; }

  const int skey = tid >> 2, schunk = tid & 3;                 // K staging
  const uint32_t kLds = (uint32_t)(uintptr_t)&Kt[0] + (uint32_t)(skey * 144 + schunk * 32);
  const int vkey = (tid & 31) * 2, vd0 = (tid >> 5) * 8;       // V staging
  const int nblocks = qt * 2 + 2;   // only key blocks <= query tile end (causal)

  for (int jb = 0; jb < nblocks; ++jb) {
    const int keybase = jb * 64;
    __syncthreads();
    {  // K: async copy rows [key][d]
      async_copy_32B(kLds, kp + (size_t)(keybase + skey) * USED_WIDTH + schunk * 16);
      // V: transpose into [d][key] with packed key-pair stores
      const bf16_t* v0 = vp + (size_t)(keybase + vkey) * USED_WIDTH + vd0;
      Frag fa, fb;
      fa.q[0] = *(const uint4*)v0;
      fb.q[0] = *(const uint4*)(v0 + USED_WIDTH);
#pragma unroll
      for (int i = 0; i < 8; ++i) {
        union { bf16_t hh[2]; uint32_t u; } pk;
        pk.hh[0] = fa.v[i];
        pk.hh[1] = fb.v[i];
        *(uint32_t*)&Vt[(vd0 + i) * 72 + vkey] = pk.u;
      }
    }
    wait_async0();
    __syncthreads();

    // S = Q @ K^T  (raw scores; scale folded into exp2)
    v8f s4[4] = {};
#pragma unroll
    for (int ct = 0; ct < 4; ++ct) {
      const char* pb = (const char*)&Kt[(ct * 16 + l15) * 72];
#pragma unroll
      for (int ks = 0; ks < 2; ++ks) {
        v16bf kf = lds_frag(pb + ks * 64 + hi * 16, 0, 32);
        s4[ct] = __builtin_amdgcn_wmma_f32_16x16x32_bf16(
            false, qf[ks], false, kf, (short)0, s4[ct], false, false);
      }
    }

    // causal mask + per-row block max
    float bmax[8];
#pragma unroll
    for (int r = 0; r < 8; ++r) bmax[r] = -3.0e38f;
#pragma unroll
    for (int ct = 0; ct < 4; ++ct) {
      const int ng = keybase + ct * 16 + l15;
#pragma unroll
      for (int r = 0; r < 8; ++r) {
        const int rg = qbase + w * 16 + hi * 8 + r;
        float vv = s4[ct][r];
        if (ng > rg) vv = -1.0e30f;
        s4[ct][r] = vv;
        bmax[r] = fmaxf(bmax[r], vv);
      }
    }
#pragma unroll
    for (int r = 0; r < 8; ++r) {
      float v0 = bmax[r];
      v0 = fmaxf(v0, __shfl_xor(v0, 1, 32));
      v0 = fmaxf(v0, __shfl_xor(v0, 2, 32));
      v0 = fmaxf(v0, __shfl_xor(v0, 4, 32));
      v0 = fmaxf(v0, __shfl_xor(v0, 8, 32));
      bmax[r] = v0;
    }
    float factor[8], rsum[8];
#pragma unroll
    for (int r = 0; r < 8; ++r) {
      const float mn = fmaxf(mrow[r], bmax[r]);
      factor[r] = exp2f((mrow[r] - mn) * SL2);
      mrow[r] = mn;
      rsum[r] = 0.f;
    }
#pragma unroll
    for (int ct = 0; ct < 4; ++ct) {
#pragma unroll
      for (int r = 0; r < 8; ++r) {
        const float p = exp2f((s4[ct][r] - mrow[r]) * SL2);
        rsum[r] += p;
        Pst[w][(hi * 8 + r) * 72 + ct * 16 + l15] = (bf16_t)p;
      }
    }
#pragma unroll
    for (int r = 0; r < 8; ++r) {
      float v0 = rsum[r];
      v0 += __shfl_xor(v0, 1, 32);
      v0 += __shfl_xor(v0, 2, 32);
      v0 += __shfl_xor(v0, 4, 32);
      v0 += __shfl_xor(v0, 8, 32);
      lrow[r] = lrow[r] * factor[r] + v0;
#pragma unroll
      for (int ct = 0; ct < 4; ++ct) o[ct][r] *= factor[r];
    }

    // same-wave cross-lane LDS RAW: P strip store -> A-fragment load
    asm volatile("s_wait_dscnt 0" ::: "memory");

    // O += P @ V
#pragma unroll
    for (int ks = 0; ks < 2; ++ks) {
      const char* pa = (const char*)&Pst[w][l15 * 72] + ks * 64 + hi * 16;
      v16bf pf = lds_frag(pa, 0, 32);
#pragma unroll
      for (int ct = 0; ct < 4; ++ct) {
        const char* pb = (const char*)&Vt[(ct * 16 + l15) * 72] + ks * 64 + hi * 16;
        v16bf vf = lds_frag(pb, 0, 32);
        o[ct] = __builtin_amdgcn_wmma_f32_16x16x32_bf16(
            false, pf, false, vf, (short)0, o[ct], false, false);
      }
    }
  }

  // epilogue: O / l -> bf16 attn_out[s][head*64 + d]
#pragma unroll
  for (int ct = 0; ct < 4; ++ct) {
    const int col = head * DIM_HEAD + ct * 16 + l15;
#pragma unroll
    for (int r = 0; r < 8; ++r) {
      const int row = qbase + w * 16 + hi * 8 + r;
      attn_out[(size_t)row * ATTN_INNER + col] = (bf16_t)(o[ct][r] / lrow[r]);
    }
  }
}

// ---------------------------------------------------------------------------
extern "C" void kernel_launch(void* const* d_in, const int* in_sizes, int n_in,
                              void* d_out, int out_size, void* d_ws, size_t ws_size,
                              hipStream_t stream) {
  (void)in_sizes; (void)n_in; (void)out_size; (void)ws_size;
  const float* x     = (const float*)d_in[0];
  const float* ln_w  = (const float*)d_in[1];
  const float* ln_b  = (const float*)d_in[2];
  const float* W_in  = (const float*)d_in[3];
  const float* W_out = (const float*)d_in[4];
  float* out = (float*)d_out;

  char* ws = (char*)d_ws;
  bf16_t* h    = (bf16_t*)ws;                                            // 8 MB
  bf16_t* resb = (bf16_t*)(ws + (size_t)8 * 1024 * 1024);                // 56 MB
  bf16_t* attn = (bf16_t*)(ws + (size_t)8 * 1024 * 1024
                              + (size_t)2 * S_LEN * USED_WIDTH);         // 8 MB

  // 1) LayerNorm + cast to bf16
  ln_kernel<<<dim3(S_LEN), dim3(256), 0, stream>>>(x, ln_w, ln_b, h);

  // 2) res[:, :14336] = h @ W_in[:, :14336]  (bf16 out)
  gemm_bf16_kernel<<<dim3(USED_WIDTH / 128, S_LEN / 128), dim3(256), 0, stream>>>(
      h, DIM, h, DIM, DIM /* a_split==K -> single source */,
      W_in, IN_WIDTH, resb, (float*)nullptr, USED_WIDTH, DIM);

  // 3) causal flash attention over q,k,v slices of res
  attn_kernel<<<dim3(NUM_HEADS, S_LEN / 128), dim3(256), 0, stream>>>(resb, attn);

  // 4) out = concat(attn_out, ffn_input) @ W_out  (f32 out)
  gemm_bf16_kernel<<<dim3(DIM / 128, S_LEN / 128), dim3(256), 0, stream>>>(
      attn, ATTN_INNER, resb + 3 * ATTN_INNER, USED_WIDTH, ATTN_INNER,
      W_out, DIM, (bf16_t*)nullptr, out, DIM, OUT_IN);
}
